// TransConv_78176994722523
// MI455X (gfx1250) — matrix operand order repaired and checked by reading, not compiled, not run on hardware
//
#include <hip/hip_runtime.h>
#include <hip/hip_bf16.h>

// ---------------------------------------------------------------------------
// Problem constants (fixed by the reference)
// ---------------------------------------------------------------------------
#define NROWS 20000
#define INC   512
#define HDIM  512
#define NHEAD 4
#define HHDIM 2048
#define NLAYR 2
#define KVSPLIT 25          // 20000 / 25 = 800 rows per split-K segment
#define KVSEG   800

typedef __attribute__((ext_vector_type(16))) _Float16 v16h;
typedef __attribute__((ext_vector_type(8)))  _Float16 v8h;
typedef __attribute__((ext_vector_type(8)))  float    v8f;
typedef __attribute__((__vector_size__(16))) int      b128_t;

#if defined(__AMDGCN__) && __has_builtin(__builtin_amdgcn_global_load_async_to_lds_b128)
#define HAVE_ASYNC_LDS 1
#else
#define HAVE_ASYNC_LDS 0
#endif

__device__ __forceinline__ void wait_async0() {
#if defined(__AMDGCN__)
#if __has_builtin(__builtin_amdgcn_s_wait_asynccnt)
  __builtin_amdgcn_s_wait_asynccnt(0);
#else
  asm volatile("s_wait_asynccnt 0" ::: "memory");
#endif
#endif
}

#if HAVE_ASYNC_LDS
// one 16-byte ASYNCcnt-tracked DMA: global -> LDS
__device__ __forceinline__ void async_copy_b128(_Float16* ldsDst,
                                                const _Float16* gSrc) {
  __builtin_amdgcn_global_load_async_to_lds_b128(
      (__attribute__((address_space(1))) b128_t*)gSrc,
      (__attribute__((address_space(3))) b128_t*)ldsDst, 0, 0);
}
#endif

// ---------------------------------------------------------------------------
// f32 -> f16 conversion (weights and input x, once)
// ---------------------------------------------------------------------------
__global__ void f32_to_f16_kernel(const float* __restrict__ s,
                                  _Float16* __restrict__ d, size_t n) {
  size_t i = (size_t)blockIdx.x * blockDim.x + threadIdx.x;
  size_t stride = (size_t)gridDim.x * blockDim.x;
  for (; i < n; i += stride) d[i] = (_Float16)s[i];
}

__global__ void fill_zero_kernel(float* __restrict__ p, size_t n) {
  size_t i = (size_t)blockIdx.x * blockDim.x + threadIdx.x;
  size_t stride = (size_t)gridDim.x * blockDim.x;
  for (; i < n; i += stride) p[i] = 0.0f;
}

// ---------------------------------------------------------------------------
// WMMA GEMM:  C[m,n] = sum_k A[m,k] * B[n,k]  (+ bias[n])
//   A: f16 [M x K]  row-major, leading dim ldA
//   B: f16 [Nn x K] row-major, leading dim ldB   (i.e. weights W[N,K])
//   C: f32 and/or f16 out (compile-time), leading dim ldC
//   gridDim.z batches (heads): per-z pointer strides in elements.
// Workgroup tile 128x128, 8 waves in 2(M)x4(N), wave tile 64x32 -> 8 WMMAs.
// K staged through double-buffered LDS via GLOBAL_LOAD_ASYNC_TO_LDS_B128
// (ASYNCcnt-tracked DMA) -> one barrier per K-slab, DMA overlapped with WMMA.
// ---------------------------------------------------------------------------
template <bool HAS_BIAS, bool OUT_F32, bool OUT_F16>
__global__ __launch_bounds__(256)
void wmma_gemm_kernel(const _Float16* __restrict__ A, long ldA, long sAz,
                      const _Float16* __restrict__ B, long ldB, long sBz,
                      const float* __restrict__ bias,
                      float* __restrict__ Cf, _Float16* __restrict__ Ch,
                      long ldC, long sCz, int M, int Nn, int K) {
  (void)Nn;
  __shared__ _Float16 sA[2][128 * 40];
  __shared__ _Float16 sB[2][128 * 40];

  const int tid  = threadIdx.x;
  const int lane = tid & 31;
  const int wave = tid >> 5;
  const int wm   = wave >> 2;   // 0..1  (64-row slab)
  const int wn   = wave & 3;    // 0..3  (32-col slab)

  const long z = blockIdx.z;
  A += z * sAz;
  B += z * sBz;
  if (OUT_F32) Cf += z * sCz;
  if (OUT_F16) Ch += z * sCz;

  const long blockM = (long)blockIdx.y * 128;
  const long blockN = (long)blockIdx.x * 128;

  v8f acc[4][2] = {};   // 8 accumulator fragments (f32)

  // cooperative tile loads: 256 threads, each moves one 16-half row segment
  const int lrow = tid >> 1;        // 0..127
  const int lseg = tid & 1;         // 0..1  (halves 0..15 / 16..31)
  long arow = blockM + lrow;
  if (arow >= M) arow = M - 1;      // clamp (stores are masked below)
  const _Float16* aptr = A + arow * ldA + lseg * 16;
  const _Float16* bptr = B + (blockN + lrow) * ldB + lseg * 16;
  const int ldsOff = lrow * 40 + lseg * 16;

  const int r    = lane & 15;
  const int hsel = lane >> 4;
  const int koff = hsel * 8;
  const int ksel = hsel * 16;

  const int nslabs = K >> 5;

#if HAVE_ASYNC_LDS
  // ---- async double-buffered pipeline ----
  {
    // prologue: DMA slab 0 into buffer 0
    async_copy_b128(&sA[0][ldsOff],     aptr);
    async_copy_b128(&sA[0][ldsOff + 8], aptr + 8);
    async_copy_b128(&sB[0][ldsOff],     bptr);
    async_copy_b128(&sB[0][ldsOff + 8], bptr + 8);
    wait_async0();
    __syncthreads();
  }
  for (int s = 0; s < nslabs; ++s) {
    const int cur = s & 1;
    if (s + 1 < nslabs) {           // kick off DMA of next slab into other buf
      aptr += 32; bptr += 32;
      const int nxt = cur ^ 1;
      async_copy_b128(&sA[nxt][ldsOff],     aptr);
      async_copy_b128(&sA[nxt][ldsOff + 8], aptr + 8);
      async_copy_b128(&sB[nxt][ldsOff],     bptr);
      async_copy_b128(&sB[nxt][ldsOff + 8], bptr + 8);
    }

    v16h afr[4];
    v16h bfr[2];
#pragma unroll
    for (int i = 0; i < 4; ++i) {
      const int row = wm * 64 + i * 16 + r;
      union { v16h v; v8h h[2]; } u;
      u.h[0] = *(const v8h*)&sA[cur][row * 40 + koff];
      u.h[1] = *(const v8h*)&sA[cur][row * 40 + 16 + koff];
      afr[i] = u.v;
    }
#pragma unroll
    for (int j = 0; j < 2; ++j) {
      const int col = wn * 32 + j * 16 + r;
      union { v16h v; v8h h[2]; } u;
      u.h[0] = *(const v8h*)&sB[cur][col * 40 + koff];
      u.h[1] = *(const v8h*)&sB[cur][col * 40 + 16 + koff];
      bfr[j] = u.v;
    }
#pragma unroll
    for (int i = 0; i < 4; ++i)
#pragma unroll
      for (int j = 0; j < 2; ++j)
        acc[i][j] = __builtin_amdgcn_wmma_f32_16x16x32_f16(
            false, afr[i], false, bfr[j], (short)0, acc[i][j], false, false);

    wait_async0();      // own DMA of next slab landed
    __syncthreads();    // everyone's DMA landed + everyone done reading cur
  }
#else
  // ---- fallback: synchronous register-staged pipeline ----
  _Float16* sAw = &sA[0][ldsOff];
  _Float16* sBw = &sB[0][ldsOff];
  for (int s = 0; s < nslabs; ++s) {
    uint4 av0 = *(const uint4*)(aptr);
    uint4 av1 = *(const uint4*)(aptr + 8);
    uint4 bv0 = *(const uint4*)(bptr);
    uint4 bv1 = *(const uint4*)(bptr + 8);
    __builtin_prefetch(aptr + 32, 0, 0);
    __builtin_prefetch(bptr + 32, 0, 0);
    aptr += 32; bptr += 32;

    __syncthreads();
    *(uint4*)(sAw)     = av0;
    *(uint4*)(sAw + 8) = av1;
    *(uint4*)(sBw)     = bv0;
    *(uint4*)(sBw + 8) = bv1;
    __syncthreads();

    v16h afr[4];
    v16h bfr[2];
#pragma unroll
    for (int i = 0; i < 4; ++i) {
      const int row = wm * 64 + i * 16 + r;
      union { v16h v; v8h h[2]; } u;
      u.h[0] = *(const v8h*)&sA[0][row * 40 + koff];
      u.h[1] = *(const v8h*)&sA[0][row * 40 + 16 + koff];
      afr[i] = u.v;
    }
#pragma unroll
    for (int j = 0; j < 2; ++j) {
      const int col = wn * 32 + j * 16 + r;
      union { v16h v; v8h h[2]; } u;
      u.h[0] = *(const v8h*)&sB[0][col * 40 + koff];
      u.h[1] = *(const v8h*)&sB[0][col * 40 + 16 + koff];
      bfr[j] = u.v;
    }
#pragma unroll
    for (int i = 0; i < 4; ++i)
#pragma unroll
      for (int j = 0; j < 2; ++j)
        acc[i][j] = __builtin_amdgcn_wmma_f32_16x16x32_f16(
            false, afr[i], false, bfr[j], (short)0, acc[i][j], false, false);
  }
#endif

  // epilogue: C layout -> lane covers N=r (+16 lanes => M+8), VGPR e -> row e
  const bool full = (blockM + 128 <= (long)M);
  if (full) {
#pragma unroll
    for (int i = 0; i < 4; ++i) {
#pragma unroll
      for (int j = 0; j < 2; ++j) {
        const long ncol = blockN + wn * 32 + j * 16 + r;
        const float bv = HAS_BIAS ? bias[ncol] : 0.0f;
#pragma unroll
        for (int e = 0; e < 8; ++e) {
          const long mrow = blockM + wm * 64 + i * 16 + (hsel ? e + 8 : e);
          const float val = acc[i][j][e] + bv;
          const long off = mrow * ldC + ncol;
          if (OUT_F32) Cf[off] = val;
          if (OUT_F16) Ch[off] = (_Float16)val;
        }
      }
    }
  } else {
#pragma unroll
    for (int i = 0; i < 4; ++i) {
#pragma unroll
      for (int j = 0; j < 2; ++j) {
        const long ncol = blockN + wn * 32 + j * 16 + r;
        const float bv = HAS_BIAS ? bias[ncol] : 0.0f;
#pragma unroll
        for (int e = 0; e < 8; ++e) {
          const long mrow = blockM + wm * 64 + i * 16 + (hsel ? e + 8 : e);
          if (mrow < M) {
            const float val = acc[i][j][e] + bv;
            const long off = mrow * ldC + ncol;
            if (OUT_F32) Cf[off] = val;
            if (OUT_F16) Ch[off] = (_Float16)val;
          }
        }
      }
    }
  }
}

// ---------------------------------------------------------------------------
// kvT[h][d][m] += sum_n v[n, h*512+d] * k[n, h*512+m]   (split-K over n)
// WMMA with per-lane strided operand gathers; f32 atomic accumulation.
// Block = 128 threads (4 waves, 2x2), block tile 64(d) x 64(m).
// ---------------------------------------------------------------------------
__global__ __launch_bounds__(128)
void kvt_kernel(const float* __restrict__ v32, const _Float16* __restrict__ k16,
                float* __restrict__ kvT) {
  const int tid  = threadIdx.x;
  const int lane = tid & 31;
  const int wave = tid >> 5;
  const int wd   = wave >> 1;
  const int wmm  = wave & 1;

  const int head = blockIdx.z / KVSPLIT;
  const int seg  = blockIdx.z % KVSPLIT;
  const int dBase = blockIdx.y * 64 + wd * 32;
  const int mBase = blockIdx.x * 64 + wmm * 32;
  const int n0base = seg * KVSEG;

  const int r    = lane & 15;
  const int hsel = lane >> 4;

  v8f acc[2][2] = {};

  for (int n0 = n0base; n0 < n0base + KVSEG; n0 += 32) {
    v16h af[2], bf[2];
#pragma unroll
    for (int i = 0; i < 2; ++i) {  // A = V^T  (rows d, K = n)
      const float* col = v32 + (size_t)head * 512 + dBase + i * 16 + r;
      union { v16h v; _Float16 e[16]; } u;
      const int kb = n0 + hsel * 16;
#pragma unroll
      for (int jj = 0; jj < 8; ++jj)
        u.e[jj] = (_Float16)col[(size_t)(kb + jj) * HHDIM];
#pragma unroll
      for (int jj = 0; jj < 8; ++jj)
        u.e[8 + jj] = (_Float16)col[(size_t)(kb + 8 + jj) * HHDIM];
      af[i] = u.v;
    }
#pragma unroll
    for (int j = 0; j < 2; ++j) {  // B = K   (K = n, cols m)
      const _Float16* col = k16 + (size_t)head * 512 + mBase + j * 16 + r;
      union { v16h v; _Float16 e[16]; } u;
      const int kb = n0 + hsel * 16;
#pragma unroll
      for (int jj = 0; jj < 16; ++jj)
        u.e[jj] = col[(size_t)(kb + jj) * HHDIM];
      bf[j] = u.v;
    }
#pragma unroll
    for (int i = 0; i < 2; ++i)
#pragma unroll
      for (int j = 0; j < 2; ++j)
        acc[i][j] = __builtin_amdgcn_wmma_f32_16x16x32_f16(
            false, af[i], false, bf[j], (short)0, acc[i][j], false, false);
  }

#pragma unroll
  for (int i = 0; i < 2; ++i)
#pragma unroll
    for (int j = 0; j < 2; ++j)
#pragma unroll
      for (int e = 0; e < 8; ++e) {
        const int d = dBase + i * 16 + (hsel ? e + 8 : e);
        const int m = mBase + j * 16 + r;
        atomicAdd(&kvT[((size_t)head * 512 + d) * 512 + m], acc[i][j][e]);
      }
}

// ---------------------------------------------------------------------------
// Column sums + per-block sum-of-squares partials (deterministic).
// src: f16 [M x Ncols]; one thread per column, coalesced row sweeps.
// ---------------------------------------------------------------------------
__global__ __launch_bounds__(256)
void colsum_ssq_kernel(const _Float16* __restrict__ src, int M, int Ncols,
                       float* __restrict__ colsum, float* __restrict__ sqpart) {
  const int col = blockIdx.x * 256 + threadIdx.x;
  float cs = 0.0f, sq = 0.0f;
  const _Float16* p = src + col;
  for (int mrow = 0; mrow < M; ++mrow) {
    const float x = (float)p[(size_t)mrow * Ncols];
    cs += x;
    sq += x * x;
  }
  if (colsum) colsum[col] = cs;
  __shared__ float red[256];
  red[threadIdx.x] = sq;
  __syncthreads();
  for (int st = 128; st > 0; st >>= 1) {
    if (threadIdx.x < st) red[threadIdx.x] += red[threadIdx.x + st];
    __syncthreads();
  }
  if (threadIdx.x == 0) sqpart[blockIdx.x] = red[0];
}

// scal layout: [0..1] spare, [2] = 1/||q||, [3] = 1/||k||
__global__ __launch_bounds__(256)
void finalize_kernel(const float* __restrict__ qpart, const float* __restrict__ kpart,
                     float* __restrict__ scal, float* __restrict__ ks_sum) {
  __shared__ float sh[2];
  if (threadIdx.x == 0) {
    float sq = 0.0f, sk = 0.0f;
    for (int i = 0; i < 8; ++i) { sq += qpart[i]; sk += kpart[i]; }
    sh[0] = rsqrtf(sq);
    sh[1] = rsqrtf(sk);
    scal[2] = sh[0];
    scal[3] = sh[1];
  }
  __syncthreads();
  const float f = sh[0] * sh[1];                 // fold 1/(||q|| ||k||)
  for (int j = threadIdx.x; j < HHDIM; j += 256) ks_sum[j] *= f;
}

__global__ void scale_kvt_kernel(const float* __restrict__ kvT,
                                 _Float16* __restrict__ kvT16,
                                 const float* __restrict__ scal, size_t n) {
  const float sk = scal[3];                      // fold 1/||k|| only
  size_t i = (size_t)blockIdx.x * blockDim.x + threadIdx.x;
  size_t stride = (size_t)gridDim.x * blockDim.x;
  for (; i < n; i += stride) kvT16[i] = (_Float16)(kvT[i] * sk);
}

// ---------------------------------------------------------------------------
// denom[n][h] = sum_m q16[n, h*512+m] * ks_sum_scaled[h*512+m] + N
// one wave per (n, head)
// ---------------------------------------------------------------------------
__global__ __launch_bounds__(256)
void denom_kernel(const _Float16* __restrict__ q16,
                  const float* __restrict__ ks_sum,
                  float* __restrict__ denom) {
  const int gwave = (blockIdx.x * 256 + threadIdx.x) >> 5;
  const int lane  = threadIdx.x & 31;
  if (gwave >= NROWS * NHEAD) return;
  const int n = gwave / NHEAD;
  const int head = gwave % NHEAD;
  const _Float16* qrow = q16 + (size_t)n * HHDIM + head * 512;
  const float* ks = ks_sum + head * 512;
  float acc = 0.0f;
  for (int m = lane; m < 512; m += 32) acc += (float)qrow[m] * ks[m];
#pragma unroll
  for (int off = 16; off > 0; off >>= 1) acc += __shfl_down(acc, off, 32);
  if (lane == 0) denom[n * NHEAD + head] = acc + (float)NROWS;
}

// ---------------------------------------------------------------------------
// LayerNorm + ReLU (after fc); one block per row (512 cols, 256 threads x2)
// ---------------------------------------------------------------------------
__global__ __launch_bounds__(256)
void ln_relu_kernel(const float* __restrict__ src, const float* __restrict__ g,
                    const float* __restrict__ b, float* __restrict__ outF,
                    _Float16* __restrict__ outH) {
  const int n = blockIdx.x, tid = threadIdx.x;
  __shared__ float red[256];
  float vals[2], lsum = 0.0f, lsq = 0.0f;
#pragma unroll
  for (int t = 0; t < 2; ++t) {
    const float x = src[(size_t)n * 512 + tid + t * 256];
    vals[t] = x; lsum += x; lsq += x * x;
  }
  red[tid] = lsum; __syncthreads();
  for (int st = 128; st > 0; st >>= 1) { if (tid < st) red[tid] += red[tid + st]; __syncthreads(); }
  const float mean = red[0] * (1.0f / 512.0f); __syncthreads();
  red[tid] = lsq; __syncthreads();
  for (int st = 128; st > 0; st >>= 1) { if (tid < st) red[tid] += red[tid + st]; __syncthreads(); }
  const float var = red[0] * (1.0f / 512.0f) - mean * mean;
  const float rstd = rsqrtf(var + 1e-5f);
#pragma unroll
  for (int t = 0; t < 2; ++t) {
    const int j = tid + t * 256;
    float o = (vals[t] - mean) * rstd * g[j] + b[j];
    o = o > 0.0f ? o : 0.0f;
    outF[(size_t)n * 512 + j] = o;
    if (outH) outH[(size_t)n * 512 + j] = (_Float16)o;
  }
}

// ---------------------------------------------------------------------------
// attn epilogue: combine heads, residual, LayerNorm, ReLU (one block per row)
//   num = s_q * attnS + N * v ;  out_h = mean_head(num / denom)
//   val = 0.5 * (out_h + h_prev) -> LN -> ReLU
// ---------------------------------------------------------------------------
__global__ __launch_bounds__(256)
void attn_epilogue_kernel(const float* __restrict__ attnS, const float* __restrict__ v32,
                          const float* __restrict__ denom, const float* __restrict__ scal,
                          const float* __restrict__ hprev,
                          const float* __restrict__ g, const float* __restrict__ b,
                          float* __restrict__ outF, _Float16* __restrict__ outH) {
  const int n = blockIdx.x, tid = threadIdx.x;
  const float s_q = scal[2];
  __shared__ float red[256];
  float vals[2], lsum = 0.0f, lsq = 0.0f;
#pragma unroll
  for (int t = 0; t < 2; ++t) {
    const int j = tid + t * 256;
    float s = 0.0f;
#pragma unroll
    for (int head = 0; head < NHEAD; ++head) {
      const size_t idx = (size_t)n * HHDIM + head * 512 + j;
      const float num = s_q * attnS[idx] + (float)NROWS * v32[idx];
      s += num / denom[n * NHEAD + head];
    }
    s *= (1.0f / (float)NHEAD);
    const float val = 0.5f * (s + hprev[(size_t)n * 512 + j]);
    vals[t] = val; lsum += val; lsq += val * val;
  }
  red[tid] = lsum; __syncthreads();
  for (int st = 128; st > 0; st >>= 1) { if (tid < st) red[tid] += red[tid + st]; __syncthreads(); }
  const float mean = red[0] * (1.0f / 512.0f); __syncthreads();
  red[tid] = lsq; __syncthreads();
  for (int st = 128; st > 0; st >>= 1) { if (tid < st) red[tid] += red[tid + st]; __syncthreads(); }
  const float var = red[0] * (1.0f / 512.0f) - mean * mean;
  const float rstd = rsqrtf(var + 1e-5f);
#pragma unroll
  for (int t = 0; t < 2; ++t) {
    const int j = tid + t * 256;
    float o = (vals[t] - mean) * rstd * g[j] + b[j];
    o = o > 0.0f ? o : 0.0f;
    outF[(size_t)n * 512 + j] = o;
    if (outH) outH[(size_t)n * 512 + j] = (_Float16)o;
  }
}

// ---------------------------------------------------------------------------
// Host launcher
// ---------------------------------------------------------------------------
extern "C" void kernel_launch(void* const* d_in, const int* in_sizes, int n_in,
                              void* d_out, int out_size, void* d_ws, size_t ws_size,
                              hipStream_t stream) {
  (void)in_sizes; (void)n_in; (void)out_size; (void)ws_size;
  const float* x    = (const float*)d_in[0];
  const float* fc_w = (const float*)d_in[1];
  const float* fc_b = (const float*)d_in[2];
  const float* ln_g = (const float*)d_in[3];
  const float* ln_b = (const float*)d_in[4];
  const float* wq   = (const float*)d_in[5];
  const float* wq_b = (const float*)d_in[6];
  const float* wk   = (const float*)d_in[7];
  const float* wk_b = (const float*)d_in[8];
  const float* wv   = (const float*)d_in[9];
  const float* wv_b = (const float*)d_in[10];

  char* base = (char*)d_ws;
  size_t off = 0;
  auto carve = [&](size_t bytes) -> char* {
    char* p = base + off;
    off = (off + bytes + 255) & ~(size_t)255;
    return p;
  };
  float*     h      = (float*)    carve((size_t)NROWS * HDIM * 4);
  _Float16*  h16    = (_Float16*) carve((size_t)NROWS * HDIM * 2);
  _Float16*  fcw16  = (_Float16*) carve((size_t)HDIM * INC * 2);
  _Float16*  wq16   = (_Float16*) carve((size_t)NLAYR * HHDIM * HDIM * 2);
  _Float16*  wk16   = (_Float16*) carve((size_t)NLAYR * HHDIM * HDIM * 2);
  _Float16*  wv16   = (_Float16*) carve((size_t)NLAYR * HHDIM * HDIM * 2);
  _Float16*  q16    = (_Float16*) carve((size_t)NROWS * HHDIM * 2);
  _Float16*  k16    = (_Float16*) carve((size_t)NROWS * HHDIM * 2);
  float*     v32    = (float*)    carve((size_t)NROWS * HHDIM * 4);
  float*     attn32 = (float*)    carve((size_t)NROWS * HHDIM * 4);
  float*     kvT32  = (float*)    carve((size_t)NHEAD * HDIM * HDIM * 4);
  _Float16*  kvT16  = (_Float16*) carve((size_t)NHEAD * HDIM * HDIM * 2);
  float*     ks_sum = (float*)    carve((size_t)HHDIM * 4);
  float*     denom  = (float*)    carve((size_t)NROWS * NHEAD * 4);
  float*     scal   = (float*)    carve(64);
  float*     qpart  = (float*)    carve(8 * 4);
  float*     kpart  = (float*)    carve(8 * 4);

  // weight + input conversions
  f32_to_f16_kernel<<<128, 256, 0, stream>>>(fc_w, fcw16, (size_t)HDIM * INC);
  f32_to_f16_kernel<<<512, 256, 0, stream>>>(wq, wq16, (size_t)NLAYR * HHDIM * HDIM);
  f32_to_f16_kernel<<<512, 256, 0, stream>>>(wk, wk16, (size_t)NLAYR * HHDIM * HDIM);
  f32_to_f16_kernel<<<512, 256, 0, stream>>>(wv, wv16, (size_t)NLAYR * HHDIM * HDIM);
  f32_to_f16_kernel<<<1024, 256, 0, stream>>>(x, h16, (size_t)NROWS * INC);

  const int mtiles = (NROWS + 127) / 128;

  // fc GEMM -> attn32 (scratch), then LN+ReLU -> h / h16
  {
    dim3 grid(HDIM / 128, mtiles, 1);
    wmma_gemm_kernel<true, true, false><<<grid, 256, 0, stream>>>(
        h16, INC, 0, fcw16, INC, 0, fc_b, attn32, nullptr, HDIM, 0,
        NROWS, HDIM, INC);
    ln_relu_kernel<<<NROWS, 256, 0, stream>>>(attn32, ln_g, ln_b, h, h16);
  }

  for (int l = 0; l < NLAYR; ++l) {
    const _Float16* Wq = wq16 + (size_t)l * HHDIM * HDIM;
    const _Float16* Wk = wk16 + (size_t)l * HHDIM * HDIM;
    const _Float16* Wv = wv16 + (size_t)l * HHDIM * HDIM;
    const float* bq = wq_b + (size_t)l * HHDIM;
    const float* bk = wk_b + (size_t)l * HHDIM;
    const float* bv = wv_b + (size_t)l * HHDIM;

    dim3 gq(HHDIM / 128, mtiles, 1);
    wmma_gemm_kernel<true, false, true><<<gq, 256, 0, stream>>>(
        h16, HDIM, 0, Wq, HDIM, 0, bq, nullptr, q16, HHDIM, 0, NROWS, HHDIM, HDIM);
    wmma_gemm_kernel<true, false, true><<<gq, 256, 0, stream>>>(
        h16, HDIM, 0, Wk, HDIM, 0, bk, nullptr, k16, HHDIM, 0, NROWS, HHDIM, HDIM);
    wmma_gemm_kernel<true, true, false><<<gq, 256, 0, stream>>>(
        h16, HDIM, 0, Wv, HDIM, 0, bv, v32, nullptr, HHDIM, 0, NROWS, HHDIM, HDIM);

    colsum_ssq_kernel<<<HHDIM / 256, 256, 0, stream>>>(q16, NROWS, HHDIM, nullptr, qpart);
    colsum_ssq_kernel<<<HHDIM / 256, 256, 0, stream>>>(k16, NROWS, HHDIM, ks_sum, kpart);
    finalize_kernel<<<1, 256, 0, stream>>>(qpart, kpart, scal, ks_sum);

    fill_zero_kernel<<<256, 256, 0, stream>>>(kvT32, (size_t)NHEAD * HDIM * HDIM);
    {
      dim3 gk(8, 8, NHEAD * KVSPLIT);
      kvt_kernel<<<gk, 128, 0, stream>>>(v32, k16, kvT32);
    }
    scale_kvt_kernel<<<256, 256, 0, stream>>>(kvT32, kvT16, scal,
                                              (size_t)NHEAD * HDIM * HDIM);

    // attention GEMM, batched over heads via gridDim.z
    {
      dim3 ga(HDIM / 128, mtiles, NHEAD);
      wmma_gemm_kernel<false, true, false><<<ga, 256, 0, stream>>>(
          q16, HHDIM, 512, kvT16, HDIM, (long)HDIM * HDIM, nullptr,
          attn32, nullptr, HHDIM, 512, NROWS, HDIM, HDIM);
    }

    denom_kernel<<<(NROWS * NHEAD + 7) / 8, 256, 0, stream>>>(q16, ks_sum, denom);

    float* outF = (l == NLAYR - 1) ? (float*)d_out : h;
    attn_epilogue_kernel<<<NROWS, 256, 0, stream>>>(
        attn32, v32, denom, scal, h,
        ln_g + (size_t)(l + 1) * HDIM, ln_b + (size_t)(l + 1) * HDIM,
        outF, h16);
  }
}
